// CharRNN_12283606468102
// MI455X (gfx1250) — matrix operand order repaired and checked by reading, not compiled, not run on hardware
//
#include <hip/hip_runtime.h>
#include <hip/hip_bf16.h>
#include <math.h>

// CharRNN miLSTM for MI455X (gfx1250, wave32, WMMA).
//
// Phases:
//  1) conv_weights : Wx/Wh (L,E,G) f32 -> transposed (L,G,E) f16 in ws
//  2) conv_sw     : softmax_w (V,E) f32 -> f16 (already B-transposed layout)
//  3) embed       : gather embedding -> h0 buffer (T,B,E) f16
//  4) milstm      : persistent single-workgroup recurrence; per-layer weights
//                   staged in 306 KB of LDS (CDNA5: 320 KB/WGP) with padded
//                   rows for bank-conflict-free ds_load_b128 fragment loads;
//                   cell state in registers; h/x double-buffered in LDS;
//                   one barrier per timestep; no VGPR spills.
//  5) logits      : (4096x128)@(128x32000) WMMA GEMM + bias, f32 out
//                   (store-bandwidth bound: 524 MB @ 23.3 TB/s ~ 22.5 us)

typedef __attribute__((ext_vector_type(16))) _Float16 v16h;
typedef __attribute__((ext_vector_type(8)))  _Float16 v8h;
typedef __attribute__((ext_vector_type(8)))  float    v8f;

#define kV 32000
#define kE 128
#define kL 2
#define kB 32
#define kT 128
#define kG 512    // 4*kE
#define kWrow 136 // padded LDS row stride in halfs (272 B -> conflict-free)

// ---- WMMA fragment loaders (layouts per CDNA5 ISA 7.12.2) ----

// A-matrix 16x32 f16 from row pointer `prow` (this lane's row m = lane&15).
// K = half*16 + {0..15} (contiguous, matching load_b_frag).
__device__ __forceinline__ v16h load_a_frag(const _Float16* prow, int kb, int half) {
  v8h lo = *(const v8h*)(prow + kb + half * 16);
  v8h hi = *(const v8h*)(prow + kb + half * 16 + 8);
  v16h r;
#pragma unroll
  for (int i = 0; i < 8; ++i) { r[i] = lo[i]; r[8 + i] = hi[i]; }
  return r;
}

// B-matrix 32x16 f16 from transposed (N, ld) storage: lane n = lane&15 holds
// K = half*16 + {0..15} -> 32 contiguous bytes, loaded as two 16-B chunks
// (row base may be only 16-B aligned with the padded stride).
__device__ __forceinline__ v16h load_b_frag(const _Float16* bt_row, int kb, int half) {
  const _Float16* p = bt_row + kb + half * 16;
  v8h lo = *(const v8h*)(p);
  v8h hi = *(const v8h*)(p + 8);
  v16h r;
#pragma unroll
  for (int i = 0; i < 8; ++i) { r[i] = lo[i]; r[8 + i] = hi[i]; }
  return r;
}

__device__ __forceinline__ float sigmoidf_(float x) { return 1.0f / (1.0f + expf(-x)); }

// ---- prep kernels ------------------------------------------------------

__global__ void conv_weights_kernel(const float* __restrict__ Wx,
                                    const float* __restrict__ Wh,
                                    _Float16* __restrict__ WxT,
                                    _Float16* __restrict__ WhT) {
  const int idx = blockIdx.x * 256 + threadIdx.x;   // (l, g, e), size L*G*E
  const int e = idx & (kE - 1);
  const int g = (idx >> 7) & (kG - 1);
  const int l = idx >> 16;                          // G*E = 65536
  WxT[idx] = (_Float16)Wx[((size_t)l * kE + e) * kG + g];
  WhT[idx] = (_Float16)Wh[((size_t)l * kE + e) * kG + g];
}

__global__ void conv_sw_kernel(const float* __restrict__ sw,
                               _Float16* __restrict__ sw16) {
  const int idx = blockIdx.x * 256 + threadIdx.x;   // V*E elements
  sw16[idx] = (_Float16)sw[idx];
}

__global__ void embed_kernel(const int* __restrict__ ids,
                             const float* __restrict__ emb,
                             _Float16* __restrict__ x) {   // x: (T,B,E) f16
  const int idx = blockIdx.x * 256 + threadIdx.x;
  const int e = idx & (kE - 1);
  const int b = (idx >> 7) & (kB - 1);
  const int t = idx >> 12;                          // kB*kE = 4096
  const int tok = ids[b * kT + t];
  x[idx] = (_Float16)emb[(size_t)tok * kE + e];
}

// ---- persistent miLSTM -------------------------------------------------
// 512 threads = 16 wave32. Wave w: mt = w&1 (batch row tile), ne = w>>1
// (E-column tile). Each wave computes the i/j/f/o gate tiles for its
// (16 rows x 16 cols) patch, so the whole cell update is register-local.
// Dynamic LDS: Wx (512x136 h) + Wh (512x136 h) + 2x h-buf + 2x x-buf.

__launch_bounds__(512, 1)
__global__ void milstm_kernel(
    const _Float16* __restrict__ WxT, const _Float16* __restrict__ WhT,
    const float* __restrict__ alpha, const float* __restrict__ beta1,
    const float* __restrict__ beta2, const float* __restrict__ bias,
    const float* __restrict__ wi, const float* __restrict__ wf,
    const float* __restrict__ wo,
    _Float16* __restrict__ h0, _Float16* __restrict__ h1) {
  extern __shared__ _Float16 smem[];
  _Float16* wxl = smem;                       // kG*kWrow halfs (136.0 KB)
  _Float16* whl = wxl + kG * kWrow;           // kG*kWrow halfs
  _Float16* shh = whl + kG * kWrow;           // 2 * kB * kWrow halfs
  _Float16* shx = shh + 2 * kB * kWrow;       // 2 * kB * kWrow halfs

  const int tid  = threadIdx.x;
  const int wave = tid >> 5;
  const int lane = tid & 31;
  const int half = lane >> 4;
  const int ln   = lane & 15;
  const int mt   = wave & 1;        // 0..1  (B = 32 -> two 16-row tiles)
  const int ne   = wave >> 1;       // 0..7  (E = 128 -> eight 16-col tiles)
  const int e    = ne * 16 + ln;    // this lane's E column
  const int arow = mt * 16 + ln;    // this lane's A-matrix row

  for (int l = 0; l < kL; ++l) {
    const _Float16* in  = (l == 0) ? h0 : h1;
    _Float16*       out = (l == 0) ? h1 : h0;

    // stage this layer's weights into LDS (padded rows: 17 int4 per row)
    {
      const int4* sx = (const int4*)(WxT + (size_t)l * kG * kE);
      const int4* sh = (const int4*)(WhT + (size_t)l * kG * kE);
      int4* dx = (int4*)wxl;
      int4* dh = (int4*)whl;
      for (int f = tid; f < kG * (kE / 8); f += 512) {  // 8192 int4 each
        const int row = f >> 4;
        const int o   = f & 15;
        dx[row * 17 + o] = sx[f];
        dh[row * 17 + o] = sh[f];
      }
    }

    float al[4], b1[4], b2[4], bb[4];
#pragma unroll
    for (int gi = 0; gi < 4; ++gi) {
      al[gi] = alpha[l * kG + gi * kE + e];
      b1[gi] = beta1[l * kG + gi * kE + e];
      b2[gi] = beta2[l * kG + gi * kE + e];
      bb[gi] = bias [l * kG + gi * kE + e];
    }
    const float wiv = wi[l * kE + e];
    const float wfv = wf[l * kE + e];
    const float wov = wo[l * kE + e];

    float creg[8];                         // cell state: 8 rows owned by lane
#pragma unroll
    for (int v = 0; v < 8; ++v) creg[v] = 0.0f;

    // init parity-0 buffers: h = 0, x = in[t=0]
#pragma unroll
    for (int v = 0; v < 8; ++v) {
      const int m = mt * 16 + v + 8 * half;
      shh[m * kWrow + e] = (_Float16)0.0f;
      shx[m * kWrow + e] = in[m * kE + e];
    }
    __syncthreads();

    for (int t = 0; t < kT; ++t) {
      const int p  = t & 1;
      const int pn = p ^ 1;
      const v8f vz = {0.f, 0.f, 0.f, 0.f, 0.f, 0.f, 0.f, 0.f};
      v8f accx[4] = {vz, vz, vz, vz};      // x@Wx tiles for i,j,f,o
      v8f acch[4] = {vz, vz, vz, vz};      // h@Wh tiles for i,j,f,o
      const _Float16* xrow = shx + p * kB * kWrow + arow * kWrow;
      const _Float16* hrow = shh + p * kB * kWrow + arow * kWrow;
#pragma unroll
      for (int kk = 0; kk < 4; ++kk) {     // K = 128 -> 4 chunks of 32
        const int kb = kk * 32;
        v16h ax = load_a_frag(xrow, kb, half);
        v16h ah = load_a_frag(hrow, kb, half);
#pragma unroll
        for (int gi = 0; gi < 4; ++gi) {
          const int col = gi * kE + ne * 16 + ln;       // column in G
          v16h bx = load_b_frag(wxl + col * kWrow, kb, half);
          accx[gi] = __builtin_amdgcn_wmma_f32_16x16x32_f16(
              false, ax, false, bx, (short)0, accx[gi], false, false);
          v16h bh = load_b_frag(whl + col * kWrow, kb, half);
          acch[gi] = __builtin_amdgcn_wmma_f32_16x16x32_f16(
              false, ah, false, bh, (short)0, acch[gi], false, false);
        }
      }
      // gates + cell update, fully register-local (C-layout: M = v + 8*half)
#pragma unroll
      for (int v = 0; v < 8; ++v) {
        const int m = mt * 16 + v + 8 * half;
        float gv[4];
#pragma unroll
        for (int gi = 0; gi < 4; ++gi) {
          const float xw  = accx[gi][v];
          const float hhv = acch[gi][v];
          gv[gi] = al[gi] * xw * hhv + b1[gi] * xw + b2[gi] * hhv + bb[gi];
        }
        const float cold = creg[v];
        const float iv = gv[0] + wiv * cold;            // input-gate peephole
        const float jv = gv[1];
        const float fv = gv[2] + wfv * cold;            // forget-gate peephole
        const float cn = sigmoidf_(fv + 1.0f) * cold + sigmoidf_(iv) * tanhf(jv);
        const float ov = gv[3] + wov * cn;              // output-gate peephole
        const float hn = sigmoidf_(ov) * tanhf(cn);
        creg[v] = cn;
        const _Float16 h16v = (_Float16)hn;
        shh[pn * kB * kWrow + m * kWrow + e] = h16v;
        out[(size_t)t * kB * kE + m * kE + e] = h16v;
        if (t + 1 < kT)
          shx[pn * kB * kWrow + m * kWrow + e] =
              in[(size_t)(t + 1) * kB * kE + m * kE + e];
      }
      __syncthreads();                     // one barrier per timestep
    }
    __threadfence();                       // make `out` visible to next layer
    __syncthreads();
  }
}

// ---- logits GEMM: (4096 x 128) @ (128 x 32000) + bias ------------------
// 8 waves/block share one M-tile (A reuse via L0); grid = (2000/8, 256).

__launch_bounds__(256, 1)
__global__ void logits_kernel(const _Float16* __restrict__ hseq,   // (T,B,E)
                              const _Float16* __restrict__ swt,    // (V,E)
                              const float* __restrict__ sb,
                              float* __restrict__ out) {
  const int tid  = threadIdx.x;
  const int wave = tid >> 5;
  const int lane = tid & 31;
  const int half = lane >> 4;
  const int ln   = lane & 15;
  const int mt   = blockIdx.y;                 // 0..255
  const int nt   = blockIdx.x * 8 + wave;      // 0..1999
  const int r    = mt * 16 + ln;               // output row = b*T + t
  const int b    = r >> 7;
  const int t    = r & (kT - 1);
  const _Float16* arow = hseq + (size_t)t * kB * kE + (size_t)b * kE;
  const int n    = nt * 16 + ln;
  const _Float16* brow = swt + (size_t)n * kE;

  v8f acc = {0.f, 0.f, 0.f, 0.f, 0.f, 0.f, 0.f, 0.f};
#pragma unroll
  for (int kk = 0; kk < 4; ++kk) {
    const int kb = kk * 32;
    v16h a  = load_a_frag(arow, kb, half);
    v16h bf = load_b_frag(brow, kb, half);
    acc = __builtin_amdgcn_wmma_f32_16x16x32_f16(
        false, a, false, bf, (short)0, acc, false, false);
  }
  const float biasn = sb[n];                   // all 8 rows share column n
#pragma unroll
  for (int v = 0; v < 8; ++v) {
    const int m = mt * 16 + v + 8 * half;
    out[(size_t)m * kV + n] = acc[v] + biasn;
  }
}

// ---- host glue ---------------------------------------------------------

extern "C" void kernel_launch(void* const* d_in, const int* in_sizes, int n_in,
                              void* d_out, int out_size, void* d_ws, size_t ws_size,
                              hipStream_t stream) {
  (void)in_sizes; (void)n_in; (void)out_size; (void)ws_size;

  const int*   ids   = (const int*)  d_in[0];
  const float* emb   = (const float*)d_in[1];
  const float* Wx    = (const float*)d_in[2];
  const float* Wh    = (const float*)d_in[3];
  const float* alpha = (const float*)d_in[4];
  const float* beta1 = (const float*)d_in[5];
  const float* beta2 = (const float*)d_in[6];
  const float* bias  = (const float*)d_in[7];
  const float* wi    = (const float*)d_in[8];
  const float* wf    = (const float*)d_in[9];
  const float* wo    = (const float*)d_in[10];
  const float* sw    = (const float*)d_in[11];
  const float* sb    = (const float*)d_in[12];
  float* out = (float*)d_out;

  // workspace layout (all sizes multiples of 256 bytes; ~10.3 MB total)
  char* ws = (char*)d_ws;
  size_t off = 0;
  _Float16* WxT  = (_Float16*)(ws + off); off += (size_t)kL * kG * kE * 2;  // 256 KB
  _Float16* WhT  = (_Float16*)(ws + off); off += (size_t)kL * kG * kE * 2;  // 256 KB
  _Float16* sw16 = (_Float16*)(ws + off); off += (size_t)kV * kE * 2;       // 8 MB
  _Float16* hb0  = (_Float16*)(ws + off); off += (size_t)kT * kB * kE * 2;  // 1 MB
  _Float16* hb1  = (_Float16*)(ws + off); off += (size_t)kT * kB * kE * 2;  // 1 MB

  conv_weights_kernel<<<(kL * kG * kE) / 256, 256, 0, stream>>>(Wx, Wh, WxT, WhT);
  conv_sw_kernel<<<(kV * kE) / 256, 256, 0, stream>>>(sw, sw16);
  embed_kernel<<<(kT * kB * kE) / 256, 256, 0, stream>>>(ids, emb, hb0);
  // layer 0: hb0 -> hb1, layer 1: hb1 -> hb0 (final hidden seq in hb0)
  // dynamic LDS: 2*(512*136) + 4*(32*136) halfs = 313,344 B (<= 320 KB/WGP)
  const size_t lstm_lds = (size_t)(2 * kG * kWrow + 4 * kB * kWrow) * sizeof(_Float16);
  milstm_kernel<<<1, 512, lstm_lds, stream>>>(WxT, WhT, alpha, beta1, beta2,
                                              bias, wi, wf, wo, hb0, hb1);
  logits_kernel<<<dim3((kV / 16) / 8, (kB * kT) / 16), 256, 0, stream>>>(
      hb0, sw16, sb, out);
}
